// DockPointNet_55688545960608
// MI455X (gfx1250) — matrix operand order repaired and checked by reading, not compiled, not run on hardware
//
#include <hip/hip_runtime.h>
#include <hip/hip_bf16.h>

#define NN 50000
#define NE 1600000

typedef __attribute__((ext_vector_type(16))) _Float16 v16h;
typedef __attribute__((ext_vector_type(8)))  _Float16 v8h;
typedef __attribute__((ext_vector_type(8)))  float    v8f;

// ---------- helpers ----------

// monotonic float -> u32 encoding for atomic max on floats
__device__ __forceinline__ unsigned fenc(float f) {
    unsigned b = __float_as_uint(f);
    return (b & 0x80000000u) ? ~b : (b | 0x80000000u);
}
__device__ __forceinline__ float fdec(unsigned u) {
    return (u & 0x80000000u) ? __uint_as_float(u ^ 0x80000000u) : __uint_as_float(~u);
}

// A fragment for v_wmma_f32_16x16x32_f16 from a row-major f16 tile [16][kdim].
// contiguous k order used here (same as load_b_frag): lane<16 holds M=lane, K=b0..b0+15 (b0=ks*32);
// lane>=16 holds K shifted by 16.
__device__ __forceinline__ v16h load_a_frag(const _Float16* base, int lane, int ks, int kdim) {
    int M = lane & 15;
    int half = (lane >> 4) & 1;
    int b0 = ks * 32 + half * 16;
    v8h lo = *(const v8h*)(base + M * kdim + b0);
    v8h hi = *(const v8h*)(base + M * kdim + b0 + 8);
    v16h a;
#pragma unroll
    for (int i = 0; i < 8; i++) { a[i] = lo[i]; a[8 + i] = hi[i]; }
    return a;
}

// B fragment from column-major f16 weights wT[col][kdim].
// contiguous k order used here (same as load_a_frag): lane<16 -> col=lane, K=ks*32 + 0..15; lane>=16 -> K=ks*32+16..31.
__device__ __forceinline__ v16h load_b_frag(const _Float16* wT, int lane, int ks, int nt, int kdim) {
    int col = nt * 16 + (lane & 15);
    int half = (lane >> 4) & 1;
    int k0 = ks * 32 + half * 16;
    return *(const v16h*)(wT + col * kdim + k0);
}

// reduce across the 16-lane column group (xor 1,2,4,8 keeps bit4 = row-half intact)
__device__ __forceinline__ float red16(float v) {
    v += __shfl_xor(v, 1, 32);
    v += __shfl_xor(v, 2, 32);
    v += __shfl_xor(v, 4, 32);
    v += __shfl_xor(v, 8, 32);
    return v;
}

// sin/cos of atan2(||a x b||, a.b) computed algebraically
__device__ __forceinline__ void angle_sc(float ax, float ay, float az,
                                         float bx, float by, float bz,
                                         float& s, float& c) {
    float cx = ay * bz - az * by;
    float cy = az * bx - ax * bz;
    float cz = ax * by - ay * bx;
    float cn = sqrtf(cx * cx + cy * cy + cz * cz);
    float d  = ax * bx + ay * by + az * bz;
    float r  = sqrtf(cn * cn + d * d);
    if (r > 0.f) { float ir = 1.f / r; s = cn * ir; c = d * ir; }
    else         { s = 0.f; c = 1.f; }
}

// ---------- kernels ----------

__global__ void init_agg_kernel(unsigned int* agg, int n) {
    int i = blockIdx.x * blockDim.x + threadIdx.x;
    if (i < n) agg[i] = 0x007FFFFFu;  // fenc(-inf)
}

__global__ void prep_weights_kernel(const float* W1, const float* W2, const float* Wg,
                                    _Float16* W1t, _Float16* W2t, _Float16* WgT) {
    int i = blockIdx.x * blockDim.x + threadIdx.x;
    if (i < 64 * 64) {                       // W1: [47,64] -> padded col-major [64 cols][64 k]
        int col = i >> 6, k = i & 63;
        W1t[col * 64 + k] = (_Float16)(k < 47 ? W1[k * 64 + col] : 0.f);
    } else if (i < 64 * 64 + 128 * 64) {     // W2: [64,128] -> [128 cols][64 k]
        int j = i - 64 * 64;
        int col = j >> 6, k = j & 63;
        W2t[col * 64 + k] = (_Float16)W2[k * 128 + col];
    } else if (i < 64 * 64 + 128 * 64 + 256 * 128) {  // Wg: [128,256] -> [256 cols][128 k]
        int m = i - 64 * 64 - 128 * 64;
        int col = m >> 7, k = m & 127;
        WgT[col * 128 + k] = (_Float16)Wg[k * 256 + col];
    }
}

__global__ __launch_bounds__(256) void edge_kernel(
    const float* __restrict__ x, const float* __restrict__ pos,
    const float* __restrict__ normal, const float* __restrict__ eattr,
    const int* __restrict__ ei,
    const float* __restrict__ b1, const float* __restrict__ g1, const float* __restrict__ be1,
    const float* __restrict__ b2, const float* __restrict__ g2, const float* __restrict__ be2,
    const _Float16* __restrict__ W1t, const _Float16* __restrict__ W2t,
    unsigned int* __restrict__ agg) {

    __shared__ _Float16 sMsg[8][16 * 64];
    __shared__ _Float16 sA2[8][16 * 64];
    __shared__ int sDst[8][16];

    int wid  = threadIdx.x >> 5;
    int lane = threadIdx.x & 31;
    int tile = blockIdx.x * 8 + wid;           // E/16 = 100000 tiles, 12500 blocks exact
    int row  = lane & 15;
    int half = lane >> 4;
    int e    = tile * 16 + row;

    _Float16* msg = sMsg[wid];
    int src = ei[e];
    int dst = ei[NE + e];

    // ---- phase 1: build 16x64 f16 message tile (layout: x[32] | ppf[7] | eattr[8] | pad) ----
    if (half == 0) {
        sDst[wid][row] = dst;
        float pix = pos[dst * 3 + 0], piy = pos[dst * 3 + 1], piz = pos[dst * 3 + 2];
        float pjx = pos[src * 3 + 0], pjy = pos[src * 3 + 1], pjz = pos[src * 3 + 2];
        float nix = normal[dst * 3 + 0], niy = normal[dst * 3 + 1], niz = normal[dst * 3 + 2];
        float njx = normal[src * 3 + 0], njy = normal[src * 3 + 1], njz = normal[src * 3 + 2];
        float dx = pjx - pix, dy = pjy - piy, dz = pjz - piz;
        float dist = sqrtf(dx * dx + dy * dy + dz * dz);
        float s1, c1, s2, c2, s3, c3;
        angle_sc(nix, niy, niz, dx, dy, dz, s1, c1);
        angle_sc(njx, njy, njz, dx, dy, dz, s2, c2);
        angle_sc(nix, niy, niz, njx, njy, njz, s3, c3);
        _Float16* mr = msg + row * 64;
        mr[32] = (_Float16)(dist * 0.125f);  // / RADIUS
        mr[33] = (_Float16)s1; mr[34] = (_Float16)c1;
        mr[35] = (_Float16)s2; mr[36] = (_Float16)c2;
        mr[37] = (_Float16)s3; mr[38] = (_Float16)c3;
#pragma unroll
        for (int k = 0; k < 8; k++) mr[39 + k] = (_Float16)eattr[e * 8 + k];
#pragma unroll
        for (int k = 47; k < 64; k++) mr[k] = (_Float16)0.f;
    } else {
        _Float16* mr = msg + row * 64;
#pragma unroll
        for (int k = 0; k < 32; k++) mr[k] = (_Float16)x[src * 32 + k];
    }
    __syncthreads();

    int nCol = lane & 15;

    // ---- stage 1 GEMM: [16x64(pad47)] x [64x64] ----
    v16h a0 = load_a_frag(msg, lane, 0, 64);
    v16h a1 = load_a_frag(msg, lane, 1, 64);
    float h1[4][8];
#pragma unroll
    for (int nt = 0; nt < 4; nt++) {
        v8f acc = {};
        acc = __builtin_amdgcn_wmma_f32_16x16x32_f16(false, a0, false,
                  load_b_frag(W1t, lane, 0, nt, 64), (short)0, acc, false, false);
        acc = __builtin_amdgcn_wmma_f32_16x16x32_f16(false, a1, false,
                  load_b_frag(W1t, lane, 1, nt, 64), (short)0, acc, false, false);
        float bb = b1[nt * 16 + nCol];
#pragma unroll
        for (int r = 0; r < 8; r++) {
            float v = acc[r] + bb;
            h1[nt][r] = v > 0.f ? v : 0.f;
        }
    }
    // LayerNorm over 64 (in-register, shuffle reduce per row)
    {
        float s[8], ss[8];
#pragma unroll
        for (int r = 0; r < 8; r++) { s[r] = 0.f; ss[r] = 0.f; }
#pragma unroll
        for (int nt = 0; nt < 4; nt++)
#pragma unroll
            for (int r = 0; r < 8; r++) { float v = h1[nt][r]; s[r] += v; ss[r] += v * v; }
#pragma unroll
        for (int r = 0; r < 8; r++) { s[r] = red16(s[r]); ss[r] = red16(ss[r]); }
        _Float16* A2 = sA2[wid];
#pragma unroll
        for (int nt = 0; nt < 4; nt++) {
            int N = nt * 16 + nCol;
            float gN = g1[N], bN = be1[N];
#pragma unroll
            for (int r = 0; r < 8; r++) {
                float mean = s[r] * (1.f / 64.f);
                float var  = ss[r] * (1.f / 64.f) - mean * mean;
                float rstd = rsqrtf(var + 1e-5f);
                float y = (h1[nt][r] - mean) * rstd * gN + bN;
                A2[(r + half * 8) * 64 + N] = (_Float16)y;
            }
        }
    }
    __syncthreads();

    // ---- stage 2 GEMM: [16x64] x [64x128] ----
    _Float16* A2 = sA2[wid];
    v16h c0 = load_a_frag(A2, lane, 0, 64);
    v16h c1 = load_a_frag(A2, lane, 1, 64);
    float h2[8][8];
#pragma unroll
    for (int nt = 0; nt < 8; nt++) {
        v8f acc = {};
        acc = __builtin_amdgcn_wmma_f32_16x16x32_f16(false, c0, false,
                  load_b_frag(W2t, lane, 0, nt, 64), (short)0, acc, false, false);
        acc = __builtin_amdgcn_wmma_f32_16x16x32_f16(false, c1, false,
                  load_b_frag(W2t, lane, 1, nt, 64), (short)0, acc, false, false);
        float bb = b2[nt * 16 + nCol];
#pragma unroll
        for (int r = 0; r < 8; r++) {
            float v = acc[r] + bb;
            h2[nt][r] = v > 0.f ? v : 0.f;
        }
    }
    // LayerNorm over 128 + encoded atomic segment-max scatter
    {
        float s[8], ss[8];
#pragma unroll
        for (int r = 0; r < 8; r++) { s[r] = 0.f; ss[r] = 0.f; }
#pragma unroll
        for (int nt = 0; nt < 8; nt++)
#pragma unroll
            for (int r = 0; r < 8; r++) { float v = h2[nt][r]; s[r] += v; ss[r] += v * v; }
#pragma unroll
        for (int r = 0; r < 8; r++) { s[r] = red16(s[r]); ss[r] = red16(ss[r]); }
        int dsts[8];
#pragma unroll
        for (int r = 0; r < 8; r++) dsts[r] = sDst[wid][r + half * 8];
#pragma unroll
        for (int nt = 0; nt < 8; nt++) {
            int N = nt * 16 + nCol;
            float gN = g2[N], bN = be2[N];
#pragma unroll
            for (int r = 0; r < 8; r++) {
                float mean = s[r] * (1.f / 128.f);
                float var  = ss[r] * (1.f / 128.f) - mean * mean;
                float rstd = rsqrtf(var + 1e-5f);
                float y = (h2[nt][r] - mean) * rstd * gN + bN;
                atomicMax(&agg[dsts[r] * 128 + N], fenc(y));
            }
        }
    }
}

__global__ __launch_bounds__(256) void node_kernel(
    const unsigned int* __restrict__ agg, const _Float16* __restrict__ WgT,
    const float* __restrict__ bg, const float* __restrict__ gg,
    const float* __restrict__ beg, float* __restrict__ out) {

    __shared__ _Float16 sAgg[8][16 * 128];
    int wid  = threadIdx.x >> 5;
    int lane = threadIdx.x & 31;
    int tile = blockIdx.x * 8 + wid;           // 3125 tiles of 16 nodes
    bool active = (tile * 16) < NN;
    int nb = tile * 16;
    _Float16* tA = sAgg[wid];

    if (active) {
        for (int idx = lane; idx < 16 * 128; idx += 32) {
            int r = idx >> 7, c = idx & 127;
            float f = fdec(agg[(nb + r) * 128 + c]);
            if (!(fabsf(f) <= 3.4e38f)) f = 0.f;   // -inf (no in-edges) / NaN -> 0
            tA[r * 128 + c] = (_Float16)f;
        }
    }
    __syncthreads();
    if (!active) return;

    int nCol = lane & 15;
    int half = lane >> 4;

    v16h aF[4];
#pragma unroll
    for (int ks = 0; ks < 4; ks++) aF[ks] = load_a_frag(tA, lane, ks, 128);

    float h[16][8];
#pragma unroll
    for (int nt = 0; nt < 16; nt++) {
        v8f acc = {};
#pragma unroll
        for (int ks = 0; ks < 4; ks++)
            acc = __builtin_amdgcn_wmma_f32_16x16x32_f16(false, aF[ks], false,
                      load_b_frag(WgT, lane, ks, nt, 128), (short)0, acc, false, false);
        float bb = bg[nt * 16 + nCol];
#pragma unroll
        for (int r = 0; r < 8; r++) {
            float v = acc[r] + bb;
            h[nt][r] = v > 0.f ? v : 0.f;
        }
    }
    float s[8], ss[8];
#pragma unroll
    for (int r = 0; r < 8; r++) { s[r] = 0.f; ss[r] = 0.f; }
#pragma unroll
    for (int nt = 0; nt < 16; nt++)
#pragma unroll
        for (int r = 0; r < 8; r++) { float v = h[nt][r]; s[r] += v; ss[r] += v * v; }
#pragma unroll
    for (int r = 0; r < 8; r++) { s[r] = red16(s[r]); ss[r] = red16(ss[r]); }
#pragma unroll
    for (int nt = 0; nt < 16; nt++) {
        int N = nt * 16 + nCol;
        float gN = gg[N], bN = beg[N];
#pragma unroll
        for (int r = 0; r < 8; r++) {
            float mean = s[r] * (1.f / 256.f);
            float var  = ss[r] * (1.f / 256.f) - mean * mean;
            float rstd = rsqrtf(var + 1e-5f);
            float y = (h[nt][r] - mean) * rstd * gN + bN;
            out[(nb + r + half * 8) * 256 + N] = y;
        }
    }
}

// ---------- launch ----------

extern "C" void kernel_launch(void* const* d_in, const int* in_sizes, int n_in,
                              void* d_out, int out_size, void* d_ws, size_t ws_size,
                              hipStream_t stream) {
    const float* x      = (const float*)d_in[0];
    const float* pos    = (const float*)d_in[1];
    const float* normal = (const float*)d_in[2];
    const float* eattr  = (const float*)d_in[3];
    const int*   ei     = (const int*)d_in[4];
    const float* W1  = (const float*)d_in[5];
    const float* b1  = (const float*)d_in[6];
    const float* g1  = (const float*)d_in[7];
    const float* be1 = (const float*)d_in[8];
    const float* W2  = (const float*)d_in[9];
    const float* b2  = (const float*)d_in[10];
    const float* g2  = (const float*)d_in[11];
    const float* be2 = (const float*)d_in[12];
    const float* Wg  = (const float*)d_in[13];
    const float* bg  = (const float*)d_in[14];
    const float* gg  = (const float*)d_in[15];
    const float* beg = (const float*)d_in[16];

    char* ws = (char*)d_ws;
    const size_t AGG_BYTES = (size_t)NN * 128 * sizeof(unsigned int);  // 25,600,000 (32B aligned)
    unsigned int* agg = (unsigned int*)ws;
    _Float16* W1t = (_Float16*)(ws + AGG_BYTES);                   // 64*64*2   = 8 KB
    _Float16* W2t = (_Float16*)(ws + AGG_BYTES + 8192);            // 128*64*2  = 16 KB
    _Float16* WgT = (_Float16*)(ws + AGG_BYTES + 8192 + 16384);    // 256*128*2 = 64 KB

    int aggN = NN * 128;
    init_agg_kernel<<<(aggN + 255) / 256, 256, 0, stream>>>(agg, aggN);

    int prepN = 64 * 64 + 128 * 64 + 256 * 128;  // 45056
    prep_weights_kernel<<<(prepN + 255) / 256, 256, 0, stream>>>(W1, W2, Wg, W1t, W2t, WgT);

    // E/16 = 100000 edge tiles, 8 waves/block -> 12500 blocks (exact, uniform)
    edge_kernel<<<12500, 256, 0, stream>>>(x, pos, normal, eattr, ei,
                                           b1, g1, be1, b2, g2, be2,
                                           W1t, W2t, agg);

    // N/16 = 3125 node tiles, 8 per block -> 391 blocks
    node_kernel<<<391, 256, 0, stream>>>(agg, WgT, bg, gg, beg, (float*)d_out);
}
